// ConvolutionHaiku_22643067585066
// MI455X (gfx1250) — matrix operand order repaired and checked by reading, not compiled, not run on hardware
//
#include <hip/hip_runtime.h>
#include <hip/hip_bf16.h>
#include <cstdint>

typedef __attribute__((ext_vector_type(16))) __bf16        v16bf;
typedef __attribute__((ext_vector_type(8)))  float         v8f;
typedef __attribute__((ext_vector_type(8)))  unsigned int  v8u;
typedef __attribute__((ext_vector_type(4)))  unsigned int  v4u;

#define GRIDXYZ 32
#define CIN     32
#define COUT    288
#define NBATCH  2
#define TAPS    125
#define NGRP    96                      // output channels per wave (6 tiles)
#define SLAB_SH (5 * NGRP * CIN)        // 15360 bf16 = 30720 B per (dx,dy) slab

__device__ __forceinline__ unsigned short f32_to_bf16(float f) {
    unsigned int u = __float_as_uint(f);
    u += 0x7FFFu + ((u >> 16) & 1u);          // round-to-nearest-even
    return (unsigned short)(u >> 16);
}

// ---------------------------------------------------------------- x -> bf16
__global__ void convert_x_bf16(const float* __restrict__ x,
                               unsigned short* __restrict__ xb, int n) {
    int i = blockIdx.x * blockDim.x + threadIdx.x;
    if (i < n) xb[i] = f32_to_bf16(x[i]);
}

// ------------------------------------------- build steerable kernel (exact f32)
// Output layout: kb[(tap*COUT + o)*CIN + u], bf16.
__global__ void build_kernel_bf16(const float* __restrict__ w0,
                                  const float* __restrict__ w1,
                                  const float* __restrict__ w2,
                                  const float* __restrict__ wl,
                                  unsigned short* __restrict__ kb) {
    int idx = blockIdx.x * blockDim.x + threadIdx.x;
    if (idx >= TAPS * COUT * CIN) return;
    int u   = idx & 31;
    int o   = (idx >> 5) % COUT;
    int tap = idx / (COUT * CIN);

    const float inv_fan = 0.17677669529663687f;   // 1/sqrt(32)
    float val;
    if (tap == 62) {                              // center voxel (2,2,2)
        val = (o < 32) ? wl[u * 32 + o] * inv_fan : 0.0f;
    } else {
        int iz = tap % 5, iy = (tap / 5) % 5, ix = tap / 25;
        float fx = ix - 2.0f, fy = iy - 2.0f, fz = iz - 2.0f;
        float r  = sqrtf(fx * fx + fy * fy + fz * fz);
        float rn = (r == 0.0f) ? 1.0f : r;
        float ux = fx / rn, uy = fy / rn, uz = fz / rn;

        int l, v, m; const float* w;
        if (o < 32)       { l = 0; v = o;             m = 0;            w = w0; }
        else if (o < 128) { l = 1; v = (o - 32) / 3;  m = (o - 32) % 3;  w = w1; }
        else              { l = 2; v = (o - 128) / 5; m = (o - 128) % 5; w = w2; }

        // radial soft-one-hot (8 basis) dotted with w_l[:, u, v]
        const float step  = 2.5f / 9.0f;
        const float cnorm = 1.14136f * 7.38905609893065f;  // 1.14136 * e^2
        float W = 0.0f;
        #pragma unroll
        for (int k = 0; k < 8; ++k) {
            float d  = (r - (float)(k + 1) * step) / step;
            float a  = d + 1.0f, bb = 1.0f - d;
            float sa = (a  > 0.0f) ? __expf(-1.0f / a)  : 0.0f;
            float sb = (bb > 0.0f) ? __expf(-1.0f / bb) : 0.0f;
            W += cnorm * sa * sb * w[k * CIN * 32 + u * 32 + v];
        }

        const float s3   = 1.7320508075688772f;
        const float s15  = 3.872983346207417f;
        const float s5h  = 1.1180339887498949f;   // sqrt(5)/2
        const float s15h = 1.9364916731037085f;   // sqrt(15)/2
        float Y;
        if (l == 0)      Y = 1.0f;
        else if (l == 1) Y = s3 * (m == 0 ? uy : (m == 1 ? uz : ux));
        else {
            switch (m) {
            case 0:  Y = s15 * ux * uy; break;
            case 1:  Y = s15 * uy * uz; break;
            case 2:  Y = s5h * (2.0f * uz * uz - ux * ux - uy * uy); break;
            case 3:  Y = s15 * ux * uz; break;
            default: Y = s15h * (ux * ux - uy * uy); break;
            }
        }
        val = W * (1.0f / 125.0f) * Y * inv_fan;
    }
    kb[(tap * COUT + o) * CIN + u] = f32_to_bf16(val);
}

// ------------------------------------------------------- implicit-GEMM conv
// Block = 8 waves sharing one output-channel group (ng): waves cover
// 4 y-values x 2 z-tiles at fixed (b, x).  B slabs staged in LDS per (dx,dy).
__global__ void __launch_bounds__(256)
conv_wmma(const unsigned short* __restrict__ xb,
          const unsigned short* __restrict__ kb,
          float* __restrict__ out) {
    __shared__ unsigned short sB[SLAB_SH];        // 30720 B

    const int tid  = threadIdx.x;
    const int lane = tid & 31;
    const int w    = tid >> 5;                    // wave in block, 0..7

    int bid = blockIdx.x;
    int ng  = bid % 3;  bid /= 3;                 // output-channel group
    int yg  = bid & 7;  bid >>= 3;                // y group of 4
    int x   = bid & 31; bid >>= 5;
    int b   = bid;

    const int y    = yg * 4 + (w >> 1);
    const int zt   = w & 1;
    const int z0   = zt * 16;
    const int n0   = ng * NGRP;
    const int m    = lane & 15;                   // row (M) / column (N) in tile
    const int half = lane >> 4;
    const int cb   = half * 8;                    // A K-chunk base (ISA A layout)

    v8f acc[6] = {};

    for (int dxi = 0; dxi < 5; ++dxi) {
        int xx = x + dxi - 2;
        if ((unsigned)xx >= (unsigned)GRIDXYZ) continue;   // block-uniform skip
        const unsigned short* xrow =
            xb + (size_t)(((b * GRIDXYZ + xx) * GRIDXYZ) ) * 0; // placeholder, set below
        (void)xrow;
        for (int dyi = 0; dyi < 5; ++dyi) {
            const int tapbase = (dxi * 5 + dyi) * 5;

            // ---- stage B slab (5 taps x 96 cols x K32 bf16) into LDS ----
            __syncthreads();                       // previous slab fully consumed
            {
                const int units = SLAB_SH / 8;     // 1920 x 16B
                for (int i = tid; i < units; i += 256) {
                    int tap5 = i / 384;            // 384 units per tap
                    int rem  = i - tap5 * 384;
                    const v4u* src = (const v4u*)
                        (kb + ((size_t)(tapbase + tap5) * COUT + n0) * CIN) + rem;
                    *((v4u*)&sB[tap5 * NGRP * CIN] + rem) = *src;   // ds_store_b128
                }
            }
            __syncthreads();

            // prefetch next slab into L2/L1 while this one is consumed
            if (!(dxi == 4 && dyi == 4)) {
                int nb = tapbase + 5;
                if (nb > TAPS - 5) nb = TAPS - 5;
                const char* np = (const char*)
                    (kb + ((size_t)nb * COUT + n0) * CIN);
                __builtin_prefetch(np + tid * 120, 0, 1);  // global_prefetch_b8
            }

            // ---- compute (per-wave validity; barriers above are uniform) ----
            int yy = y + dyi - 2;
            if ((unsigned)yy < (unsigned)GRIDXYZ) {
                const unsigned short* xr =
                    xb + (size_t)(((b * GRIDXYZ + xx) * GRIDXYZ + yy) * GRIDXYZ) * CIN;
                #pragma unroll
                for (int dzi = 0; dzi < 5; ++dzi) {
                    // A fragment: 16 z-rows x K=32 bf16, ISA 16-bit A layout
                    int  zz  = z0 + m + dzi - 2;
                    bool inz = (unsigned)zz < (unsigned)GRIDXYZ;
                    int  zc  = inz ? zz : 0;                  // clamped address
                    const unsigned short* ap = xr + zc * CIN + cb;
                    v4u alo = *(const v4u*)(ap);              // K cb..cb+7
                    v4u ahi = *(const v4u*)(ap + 16);         // K cb+16..cb+23
                    if (!inz) { alo = (v4u)0u; ahi = (v4u)0u; }  // zero pad (cndmask)
                    v8u au; au.lo = alo; au.hi = ahi;
                    v16bf afrag = __builtin_bit_cast(v16bf, au);

                    const unsigned short* bls =
                        &sB[dzi * NGRP * CIN + half * 8];
                    #pragma unroll
                    for (int j = 0; j < 6; ++j) {
                        // B fragment from LDS: column c = j*16+m, K half*8..+7 and 16+half*8..+7
                        const unsigned short* bp = bls + (j * 16 + m) * CIN;
                        v8u bu;
                        bu.lo = *(const v4u*)(bp);            // ds_load_b128
                        bu.hi = *(const v4u*)(bp + 16);
                        v16bf bfrag = __builtin_bit_cast(v16bf, bu);
                        acc[j] = __builtin_amdgcn_wmma_f32_16x16x32_bf16(
                            false, afrag, false, bfrag, (short)0, acc[j],
                            false, false);
                    }
                }
            }
        }
    }

    // ---- store: C/D layout -> VGPR r holds M = half*8 + r, lane&15 = N ----
    float* obase = out + (size_t)(((b * GRIDXYZ + x) * GRIDXYZ + y) * GRIDXYZ) * COUT;
    #pragma unroll
    for (int j = 0; j < 6; ++j) {
        int n = n0 + j * 16 + m;
        #pragma unroll
        for (int r = 0; r < 8; ++r) {
            int z = z0 + half * 8 + r;
            obase[(size_t)z * COUT + n] = acc[j][r];
        }
    }
}

extern "C" void kernel_launch(void* const* d_in, const int* in_sizes, int n_in,
                              void* d_out, int out_size, void* d_ws, size_t ws_size,
                              hipStream_t stream) {
    const float* x  = (const float*)d_in[0];
    const float* w0 = (const float*)d_in[1];
    const float* w1 = (const float*)d_in[2];
    const float* w2 = (const float*)d_in[3];
    const float* wl = (const float*)d_in[4];

    const int nx = NBATCH * GRIDXYZ * GRIDXYZ * GRIDXYZ * CIN;   // 2,097,152
    const int nk = TAPS * COUT * CIN;                            // 1,152,000

    unsigned short* xb = (unsigned short*)d_ws;
    unsigned short* kb = (unsigned short*)((char*)d_ws + (size_t)nx * sizeof(unsigned short));

    convert_x_bf16<<<(nx + 255) / 256, 256, 0, stream>>>(x, xb, nx);
    build_kernel_bf16<<<(nk + 255) / 256, 256, 0, stream>>>(w0, w1, w2, wl, kb);

    const int blocks = NBATCH * GRIDXYZ * 8 * 3;                 // 1536 blocks
    conv_wmma<<<blocks, 256, 0, stream>>>(xb, kb, (float*)d_out);
}
